// SimpleMACUnit_21577915695721
// MI455X (gfx1250) — matrix operand order repaired and checked by reading, not compiled, not run on hardware
//
#include <hip/hip_runtime.h>
#include <hip/hip_bf16.h>
#include <stddef.h>

typedef __bf16 bf16_t;
typedef __bf16 v16bf __attribute__((ext_vector_type(16)));
typedef float  v8f   __attribute__((ext_vector_type(8)));

#define DIMD 512
#define BSZ  128
#define LCTX 512
#define NUU  32
#define NSTEP 12

// ------------------------- bf16 WMMA GEMM -------------------------
// C(MxN) = A(MxK,bf16,row-major) @ B(KxN) + bias, B supplied PRE-TRANSPOSED
// as Bt(NxK). block: 128 threads (4 waves), tile 64x128, K-slab 32,
// double-buffered LDS. Prefetch uses SIX NAMED float4 scalars (no arrays, no
// lambdas -> guaranteed register residency, no scratch alloca) and
// strength-reduced global pointers. waves_per_eu(1) lifts the VGPR cap;
// occupancy is LDS-bound, not register-bound.
//
// Epilogue MODEs:
//  0: fOut[m*N+n] = v                                   (cq)
//  1: bfOut[m*2N + N + n] = bf16(v)                     (pa -> A_cq high half)
//  2: bfOut[m*2N + n]     = bf16(v * aux[m*N+n])        (mem_p*know_p -> A_cat low)
//  3: s = v*ctrl[(m/NU)*N+n]; fOut[m*N+n]=s; bfOut[m*N+n]=bf16(s)  (cat)
//  4: fOut[m*N+n]=v; bfOut[m*2N + N + n]=bf16(v)        (know_p f32 + A_cat high)
#define BM 64
#define BN 128
#define BKK 32
#define APAD 40
#define BPAD 40

template <int MODE>
__global__ void __launch_bounds__(128)
__attribute__((amdgpu_waves_per_eu(1)))
k_gemm(const bf16_t* __restrict__ A, const bf16_t* __restrict__ Bt,
       const float* __restrict__ bias, const float* __restrict__ aux,
       const float* __restrict__ ctrl, float* __restrict__ fOut,
       bf16_t* __restrict__ bfOut, int M, int N, int K)
{
  __shared__ bf16_t sA[2][BM * APAD];   // [m][k]
  __shared__ bf16_t sB[2][BN * BPAD];   // [n][k]

  const int tid  = threadIdx.x;
  const int wv   = tid >> 5;
  const int lane = tid & 31;
  const int hf   = lane >> 4;
  const int l16  = lane & 15;

  const int m0 = blockIdx.x * BM;
  const int n0 = blockIdx.y * BN;
  const int mw = (wv & 1) * 32;
  const int nw = (wv >> 1) * 64;

  // staging geometry: row0 = tid>>2, extra slices are +32/+64/+96 rows; same chunk
  const int row0 = tid >> 2;
  const int ch   = (tid & 3) * 8;

  const bf16_t* gA0 = A  + (size_t)(m0 + row0) * K + ch;
  const bf16_t* gA1 = gA0 + (size_t)32 * K;
  const bf16_t* gB0 = Bt + (size_t)(n0 + row0) * K + ch;
  const bf16_t* gB1 = gB0 + (size_t)32 * K;
  const bf16_t* gB2 = gB0 + (size_t)64 * K;
  const bf16_t* gB3 = gB0 + (size_t)96 * K;

  const int sa0 = row0 * APAD + ch, sa1 = sa0 + 32 * APAD;
  const int sb0 = row0 * BPAD + ch, sb1 = sb0 + 32 * BPAD,
            sb2 = sb0 + 64 * BPAD, sb3 = sb0 + 96 * BPAD;

  v8f acc[2][4] = {};
  float4 ra0, ra1, rb0, rb1, rb2, rb3;   // named scalars: always registers

#define GLOADS()                                                    \
  do {                                                              \
    ra0 = *(const float4*)gA0;  ra1 = *(const float4*)gA1;          \
    rb0 = *(const float4*)gB0;  rb1 = *(const float4*)gB1;          \
    rb2 = *(const float4*)gB2;  rb3 = *(const float4*)gB3;          \
    gA0 += BKK; gA1 += BKK; gB0 += BKK;                             \
    gB1 += BKK; gB2 += BKK; gB3 += BKK;                             \
  } while (0)

#define LSTORES(buf)                                                \
  do {                                                              \
    bf16_t* pa = sA[buf]; bf16_t* pb = sB[buf];                     \
    *(float4*)(pa + sa0) = ra0;  *(float4*)(pa + sa1) = ra1;        \
    *(float4*)(pb + sb0) = rb0;  *(float4*)(pb + sb1) = rb1;        \
    *(float4*)(pb + sb2) = rb2;  *(float4*)(pb + sb3) = rb3;        \
  } while (0)

  const int niter = K / BKK;
  GLOADS();
  LSTORES(0);
  __syncthreads();

  for (int i = 0; i < niter; ++i) {
    const int cur = i & 1;
    const bool has_next = (i + 1 < niter);
    if (has_next) GLOADS();     // issue loads; no wait until after WMMAs

    v16bf afr[2];
#pragma unroll
    for (int am = 0; am < 2; ++am) {
      const bf16_t* p0 = sA[cur] + (mw + am * 16 + l16) * APAD + hf * 16;
      __builtin_memcpy(&afr[am], p0, 16);
      __builtin_memcpy(((char*)&afr[am]) + 16, p0 + 8, 16);
    }
#pragma unroll
    for (int ns = 0; ns < 4; ++ns) {
      v16bf bfr;
      const bf16_t* pb = sB[cur] + (nw + ns * 16 + l16) * BPAD + hf * 16;
      __builtin_memcpy(&bfr, pb, 32);
#pragma unroll
      for (int am = 0; am < 2; ++am)
        acc[am][ns] = __builtin_amdgcn_wmma_f32_16x16x32_bf16(
            false, afr[am], false, bfr, (short)0, acc[am][ns], false, false);
    }

    if (has_next) LSTORES(1 - cur);   // loadcnt waits land here, after WMMAs
    __syncthreads();
  }
#undef GLOADS
#undef LSTORES

#pragma unroll
  for (int am = 0; am < 2; ++am) {
#pragma unroll
    for (int ns = 0; ns < 4; ++ns) {
      const int n = n0 + nw + ns * 16 + l16;
      const float bv = bias[n];
#pragma unroll
      for (int r = 0; r < 8; ++r) {
        const int m = m0 + mw + am * 16 + hf * 8 + r;
        const float v = acc[am][ns][r] + bv;
        if (MODE == 0) {
          fOut[(size_t)m * N + n] = v;
        } else if (MODE == 1) {
          bfOut[(size_t)m * 2 * N + N + n] = (bf16_t)v;
        } else if (MODE == 2) {
          bfOut[(size_t)m * 2 * N + n] = (bf16_t)(v * aux[(size_t)m * N + n]);
        } else if (MODE == 3) {
          const float s = v * ctrl[(size_t)(m >> 5) * N + n];   // b = m/NU (NU=32)
          fOut[(size_t)m * N + n] = s;
          bfOut[(size_t)m * N + n] = (bf16_t)s;
        } else {  // MODE == 4
          fOut[(size_t)m * N + n] = v;
          bfOut[(size_t)m * 2 * N + N + n] = (bf16_t)v;
        }
      }
    }
  }
}

// ------------------------- converters / init -------------------------
__global__ void k_cvt_bf16(const float* __restrict__ s, bf16_t* __restrict__ d, int n) {
  int i = blockIdx.x * 256 + threadIdx.x;
  if (i < n) d[i] = (bf16_t)s[i];
}

// dst(NxK) = bf16(src(KxN))^T ; one matrix slice per blockIdx.y
__global__ void k_cvt_bf16_T(const float* __restrict__ src, bf16_t* __restrict__ dst,
                             int K, int N) {
  int i = blockIdx.x * 256 + threadIdx.x;         // i = k*N + n
  size_t base = (size_t)blockIdx.y * K * N;
  int k = i / N, n = i - k * N;
  dst[base + (size_t)n * K + k] = (bf16_t)src[base + i];
}

// A_cq low half <- bf16(control_0) broadcast
__global__ void k_init_acq_lo(const float* __restrict__ c0, bf16_t* __restrict__ Acq) {
  int i = blockIdx.x * 256 + threadIdx.x;         // B*D
  Acq[(size_t)(i >> 9) * 1024 + (i & (DIMD - 1))] = (bf16_t)c0[i & (DIMD - 1)];
}

__global__ void k_init_memA(const float* __restrict__ m0, bf16_t* __restrict__ memA) {
  int i = blockIdx.x * 256 + threadIdx.x;         // B*NU*D
  memA[i] = (bf16_t)m0[i & (DIMD - 1)];
}

__global__ void k_copy_f32(const float* __restrict__ s, float* __restrict__ d) {
  int i = blockIdx.x * 256 + threadIdx.x;
  d[i] = s[i];
}

// ------------------------- attention (control unit) -------------------------
__global__ void __launch_bounds__(256)
k_attention(const float* __restrict__ cq, const float* __restrict__ W_cattn,
            const float* __restrict__ b_cattn, const float* __restrict__ context,
            float* __restrict__ new_control, bf16_t* __restrict__ AcqLo)
{
  __shared__ float sv[DIMD];
  __shared__ float sc[LCTX];
  __shared__ float sred[8];

  const int b = blockIdx.x;
  const int tid = threadIdx.x;
  const int wv = tid >> 5, lane = tid & 31;
  const float* ctx = context + (size_t)b * LCTX * DIMD;

  for (int d = tid; d < DIMD; d += 256) sv[d] = cq[(size_t)b * DIMD + d] * W_cattn[d];
  __syncthreads();

  for (int l = wv; l < LCTX; l += 8) {
    const float* row = ctx + (size_t)l * DIMD;
    float s = 0.f;
    for (int d = lane * 4; d < DIMD; d += 128) {
      float4 x = *(const float4*)(row + d);
      s += x.x * sv[d] + x.y * sv[d + 1] + x.z * sv[d + 2] + x.w * sv[d + 3];
    }
    for (int off = 16; off > 0; off >>= 1) s += __shfl_down(s, off, 32);
    if (lane == 0) sc[l] = s + b_cattn[0];
  }
  __syncthreads();

  float mx = -3.4e38f;
  for (int l = tid; l < LCTX; l += 256) mx = fmaxf(mx, sc[l]);
  for (int off = 16; off > 0; off >>= 1) mx = fmaxf(mx, __shfl_down(mx, off, 32));
  if (lane == 0) sred[wv] = mx;
  __syncthreads();
  if (tid == 0) { float m2 = sred[0]; for (int w = 1; w < 8; ++w) m2 = fmaxf(m2, sred[w]); sred[0] = m2; }
  __syncthreads();
  mx = sred[0];
  __syncthreads();

  float sum = 0.f;
  for (int l = tid; l < LCTX; l += 256) { float e = __expf(sc[l] - mx); sc[l] = e; sum += e; }
  for (int off = 16; off > 0; off >>= 1) sum += __shfl_down(sum, off, 32);
  if (lane == 0) sred[wv] = sum;
  __syncthreads();
  if (tid == 0) { float s2 = 0.f; for (int w = 0; w < 8; ++w) s2 += sred[w]; sred[0] = s2; }
  __syncthreads();
  const float inv = 1.0f / sred[0];
  for (int l = tid; l < LCTX; l += 256) sc[l] *= inv;
  __syncthreads();

  for (int d = tid; d < DIMD; d += 256) {
    float a = 0.f;
    const float* cp = ctx + d;
    for (int l = 0; l < LCTX; ++l) a += sc[l] * cp[(size_t)l * DIMD];
    new_control[(size_t)b * DIMD + d] = a;
    AcqLo[(size_t)b * 1024 + d] = (bf16_t)a;   // control for next step's cq GEMM
  }
}

// ------------------------- clip attention over units -------------------------
__global__ void __launch_bounds__(256)
k_clip(const float* __restrict__ mem, const float* __restrict__ W_lin,
       const float* __restrict__ b_lin, const unsigned char* __restrict__ masks,
       float* __restrict__ out)
{
  __shared__ float slog[NUU];
  const int b = blockIdx.x;
  const int tid = threadIdx.x;
  const int wv = tid >> 5, lane = tid & 31;

  for (int u = wv; u < NUU; u += 8) {
    const float* row = mem + ((size_t)b * NUU + u) * DIMD;
    float s = 0.f;
    for (int d = lane * 4; d < DIMD; d += 128) {
      float4 x = *(const float4*)(row + d);
      s += x.x * W_lin[d] + x.y * W_lin[d + 1] + x.z * W_lin[d + 2] + x.w * W_lin[d + 3];
    }
    for (int off = 16; off > 0; off >>= 1) s += __shfl_down(s, off, 32);
    if (lane == 0) slog[u] = masks[(size_t)b * NUU + u] ? (s + b_lin[0]) : -1e9f;
  }
  __syncthreads();

  if (wv == 0) {  // NU == 32 == wave32
    float x = slog[lane];
    float m = x;
    for (int off = 16; off > 0; off >>= 1) m = fmaxf(m, __shfl_down(m, off, 32));
    m = __shfl(m, 0, 32);
    float e = __expf(x - m);
    float s = e;
    for (int off = 16; off > 0; off >>= 1) s += __shfl_down(s, off, 32);
    s = __shfl(s, 0, 32);
    out[(size_t)b * NUU + lane] = e / s;
  }
}

// ------------------------- host orchestration -------------------------
extern "C" void kernel_launch(void* const* d_in, const int* in_sizes, int n_in,
                              void* d_out, int out_size, void* d_ws, size_t ws_size,
                              hipStream_t stream)
{
  (void)in_sizes; (void)n_in; (void)out_size; (void)ws_size;

  const float* context   = (const float*)d_in[0];
  const float* question  = (const float*)d_in[1];
  const float* knowledge = (const float*)d_in[2];
  const unsigned char* masks = (const unsigned char*)d_in[3];
  const float* W_pos    = (const float*)d_in[5];
  const float* b_pos    = (const float*)d_in[6];
  const float* W_cq     = (const float*)d_in[7];
  const float* b_cq     = (const float*)d_in[8];
  const float* W_cattn  = (const float*)d_in[9];
  const float* b_cattn  = (const float*)d_in[10];
  const float* W_mem    = (const float*)d_in[11];
  const float* b_mem    = (const float*)d_in[12];
  const float* W_know   = (const float*)d_in[13];
  const float* b_know   = (const float*)d_in[14];
  const float* W_concat = (const float*)d_in[15];
  const float* b_concat = (const float*)d_in[16];
  const float* W_lin    = (const float*)d_in[17];
  const float* b_lin    = (const float*)d_in[18];
  const float* control0 = (const float*)d_in[19];
  const float* mem0     = (const float*)d_in[20];

  const size_t NE1 = (size_t)BSZ * DIMD;          // 65536
  const size_t NE2 = (size_t)BSZ * NUU * DIMD;    // 2097152

  float* out_memory   = (float*)d_out;
  float* out_clip     = out_memory + NE2;
  float* out_memories = out_clip + (size_t)BSZ * NUU;

  char* ws = (char*)d_ws;
  size_t off = 0;
  auto take = [&](size_t bytes) -> char* {
    char* p = ws + off; off += (bytes + 255) & ~(size_t)255; return p;
  };
  bf16_t* bfWposT    = (bf16_t*)take((size_t)NSTEP * DIMD * DIMD * 2);  // W_pos[t]^T
  bf16_t* bfWcqT     = (bf16_t*)take((size_t)2 * DIMD * DIMD * 2);      // (512 x 1024)
  bf16_t* bfWmemT    = (bf16_t*)take((size_t)DIMD * DIMD * 2);
  bf16_t* bfWknowT   = (bf16_t*)take((size_t)DIMD * DIMD * 2);
  bf16_t* bfWconcatT = (bf16_t*)take((size_t)2 * DIMD * DIMD * 2);
  bf16_t* bfQ        = (bf16_t*)take(NE1 * 2);
  bf16_t* bfKnow     = (bf16_t*)take(NE2 * 2);
  bf16_t* bfAcq      = (bf16_t*)take(NE1 * 2 * 2);   // (B, 2D): [ctrl | pa]
  float*  fCq        = (float*)take(NE1 * 4);
  float*  fCtrl      = (float*)take(NE1 * 4);
  float*  fKnowP     = (float*)take(NE2 * 4);
  bf16_t* bfAcat     = (bf16_t*)take(NE2 * 2 * 2);   // (B*NU, 2D): [mem_p*know_p | know_p]
  bf16_t* bfMemA     = (bf16_t*)take(NE2 * 2);       // current memory, bf16

  // one-time conversions; weights stored TRANSPOSED so GEMM B-staging is contiguous
  k_cvt_bf16_T<<<dim3(DIMD * DIMD / 256, NSTEP), 256, 0, stream>>>(W_pos, bfWposT, DIMD, DIMD);
  k_cvt_bf16_T<<<dim3(2 * DIMD * DIMD / 256, 1), 256, 0, stream>>>(W_cq, bfWcqT, 2 * DIMD, DIMD);
  k_cvt_bf16_T<<<dim3(DIMD * DIMD / 256, 1), 256, 0, stream>>>(W_mem, bfWmemT, DIMD, DIMD);
  k_cvt_bf16_T<<<dim3(DIMD * DIMD / 256, 1), 256, 0, stream>>>(W_know, bfWknowT, DIMD, DIMD);
  k_cvt_bf16_T<<<dim3(2 * DIMD * DIMD / 256, 1), 256, 0, stream>>>(W_concat, bfWconcatT, 2 * DIMD, DIMD);
  k_cvt_bf16<<<(int)NE1 / 256, 256, 0, stream>>>(question, bfQ, (int)NE1);
  k_cvt_bf16<<<(int)NE2 / 256, 256, 0, stream>>>(knowledge, bfKnow, (int)NE2);

  k_init_acq_lo<<<NE1 / 256, 256, 0, stream>>>(control0, bfAcq);
  k_init_memA  <<<NE2 / 256, 256, 0, stream>>>(mem0, bfMemA);

  const dim3 blk(128);
  const dim3 gSmall(BSZ / BM, DIMD / BN);          // (2,4)  M=128
  const dim3 gBig((BSZ * NUU) / BM, DIMD / BN);    // (64,4) M=4096

  // hoisted: know_p = knowledge @ W_know + b_know  -> f32 + A_cat high half (bf16)
  k_gemm<4><<<gBig, blk, 0, stream>>>(bfKnow, bfWknowT, b_know, nullptr, nullptr,
                                      fKnowP, bfAcat, BSZ * NUU, DIMD, DIMD);

  for (int t = 0; t < NSTEP; ++t) {
    // pa = question @ W_pos[t] + b_pos[t]  -> A_cq high half (bf16)
    k_gemm<1><<<gSmall, blk, 0, stream>>>(bfQ, bfWposT + (size_t)t * DIMD * DIMD,
                                          b_pos + (size_t)t * DIMD, nullptr, nullptr,
                                          nullptr, bfAcq, BSZ, DIMD, DIMD);
    // cq = [control_prev | pa] @ W_cq + b_cq
    k_gemm<0><<<gSmall, blk, 0, stream>>>(bfAcq, bfWcqT, b_cq, nullptr, nullptr,
                                          fCq, nullptr, BSZ, DIMD, 2 * DIMD);
    // attention -> new_control (f32) + A_cq low half for next step (bf16)
    k_attention<<<BSZ, 256, 0, stream>>>(fCq, W_cattn, b_cattn, context, fCtrl, bfAcq);
    // mem_p = memory_prev @ W_mem + b_mem ; A_cat low = bf16(mem_p * know_p)
    k_gemm<2><<<gBig, blk, 0, stream>>>(bfMemA, bfWmemT, b_mem, fKnowP, nullptr,
                                        nullptr, bfAcat, BSZ * NUU, DIMD, DIMD);
    // cat = A_cat @ W_concat + b_concat, scaled by new_control
    //   -> memories[t] (f32) and next-step memory (bf16)
    float* slice = out_memories + (size_t)t * NE2;
    k_gemm<3><<<gBig, blk, 0, stream>>>(bfAcat, bfWconcatT, b_concat, nullptr, fCtrl,
                                        slice, bfMemA, BSZ * NUU, DIMD, 2 * DIMD);
  }

  k_copy_f32<<<NE2 / 256, 256, 0, stream>>>(out_memories + (size_t)(NSTEP - 1) * NE2,
                                            out_memory);
  k_clip<<<BSZ, 256, 0, stream>>>(out_memory, W_lin, b_lin, masks, out_clip);
}
